// LinearAttention_45999099740165
// MI455X (gfx1250) — matrix operand order repaired and checked by reading, not compiled, not run on hardware
//
#include <hip/hip_runtime.h>

// ---------------------------------------------------------------------------
// LinearAttention (dual-softmax linear attention with RoPE-over-heads and
// memory KV) for MI455X / gfx1250.
//
// Inputs (setup_inputs order):
//   d_in[0] x      : (4, 8192, 1024) f32
//   d_in[1] mask   : (4, 8192)       i32
//   d_in[2] W_qkv  : (3072, 1024)    f32
//   d_in[3] mem_kv : (2, 16, 4, 64)  f32
// Output: (4, 8192, 1024) f32
//
// QKV channel mapping: c = h*192 + d*3 + comp (comp: 0=q, 1=k, 2=v)
//
// Pipeline:
//   K0  cvt_bf16      : x, W -> bf16 copies in workspace (one streaming pass)
//   K1  qkv_gemm_wmma : TDM (tensor_load_to_lds) double-buffered LDS tiles,
//                       v_wmma_f32_16x16x32_bf16, f32 accumulate
//   K2  q_rope_softmax: RoPE + feature softmax on q (in-place)
//   K3  k_stats       : per-(b,h,d) max / sum-exp over 8196 sequence rows
//   K4  ctx_kernel    : ctx = softmax_k^T * v  (64x64 per (b,h))
//   K5  out_kernel    : out = (q_sm @ ctx) * mask
// ---------------------------------------------------------------------------

#define B_      4
#define L_      8192
#define DIN_    1024
#define H_      16
#define DH_     64
#define C3_     3072          // 3 * d_model
#define NMEM_   4
#define NSEQ_   (L_ + NMEM_)  // 8196

typedef __attribute__((ext_vector_type(8)))  float        v8f;
typedef __attribute__((ext_vector_type(8)))  __bf16       v8bf;
typedef __attribute__((ext_vector_type(16))) __bf16       v16bf;
typedef __attribute__((ext_vector_type(4)))  unsigned int u32x4;
typedef __attribute__((ext_vector_type(4)))  int          i32x4;
typedef __attribute__((ext_vector_type(8)))  int          i32x8;

// ln(10000)/32 : theta_j = exp(-j * this)
#define ROPE_LN_SCALE 0.28782313662425572f

__device__ __forceinline__ unsigned short f2bf(float f) {
  unsigned int u = __float_as_uint(f);
  u += 0x7FFFu + ((u >> 16) & 1u);     // round-to-nearest-even
  return (unsigned short)(u >> 16);
}

// ===========================================================================
// K0: f32 -> bf16 conversion pass (8 elements / thread, 16 B packed stores)
// ===========================================================================
__global__ __launch_bounds__(256) void cvt_bf16(
    const float* __restrict__ src, unsigned short* __restrict__ dst)
{
  const size_t i = ((size_t)blockIdx.x * 256 + threadIdx.x) * 8;
  float4 a = ((const float4*)(src + i))[0];
  float4 b = ((const float4*)(src + i))[1];
  uint4 o;
  o.x = (unsigned)f2bf(a.x) | ((unsigned)f2bf(a.y) << 16);
  o.y = (unsigned)f2bf(a.z) | ((unsigned)f2bf(a.w) << 16);
  o.z = (unsigned)f2bf(b.x) | ((unsigned)f2bf(b.y) << 16);
  o.w = (unsigned)f2bf(b.z) | ((unsigned)f2bf(b.w) << 16);
  *(uint4*)(dst + i) = o;
}

// ===========================================================================
// Tensor Data Mover helper: 2D tile load (global bf16 -> LDS), per-block
// uniform descriptor.  Group1 is prebuilt; group0 carries lds/global addr.
//   group0: w0 = count=1 ; w1 = lds_addr ; w2 = gaddr[31:0] ;
//           w3 = gaddr[56:32] | type(2)<<30
// ===========================================================================
#if __has_include(<hip/amd_detail/amd_gfx1250_TDM.h>)
#define TDM_6ARG 1   // therock headers => 6-arg builtin toolchain
#endif

__device__ __forceinline__ void tdm_load_2d(unsigned lds_off, const void* gptr,
                                            i32x8 g1) {
  const unsigned long long ga = (unsigned long long)(size_t)gptr;
  u32x4 g0 = { 1u, lds_off, (unsigned)ga,
               ((unsigned)((ga >> 32) & 0x01FFFFFFu)) | (2u << 30) };
#ifdef TDM_6ARG
  __builtin_amdgcn_tensor_load_to_lds(g0, g1, (i32x4)0, (i32x4)0, (i32x8)0, 0);
#else
  __builtin_amdgcn_tensor_load_to_lds(g0, g1, (i32x4)0, (i32x4)0, 0);
#endif
}

// group1 for a (rows x 1024) bf16 tensor, 128x32 tile, +16B row pad in LDS:
//   w0: data_size=1(2B)<<16 | pad_enable<<20 | pad_interval=3(16 dw)<<22
//       | pad_amount=3(4 dw)<<25
//   w1: tensor_dim0(=1024) << 16
//   w2: tensor_dim1[15:0] << 16
//   w3: tensor_dim1[31:16] | tile_dim0(=32) << 16
//   w4: tile_dim1(=128)          w5: tensor_dim0_stride(=1024)
__device__ __forceinline__ i32x8 tdm_group1(int nrows) {
  const int w0 = (1 << 16) | (1 << 20) | (3 << 22) | (3 << 25);
  i32x8 g1 = { w0, (int)(DIN_ << 16), (int)((nrows & 0xFFFF) << 16),
               (int)((nrows >> 16) & 0xFFFF) | (32 << 16),
               128, DIN_, 0, 0 };
  return g1;
}

// ===========================================================================
// K1: QKV = Xb * Wb^T with bf16 WMMA, f32 accumulate.
//     Block tile 128(M) x 128(N), K-panel 32, TDM double-buffered staging.
//     8 waves: 2(M) x 4(N), each wave computes 4x2 tiles of 16x16.
// ===========================================================================
#define TM   128
#define TN   128
#define TK   32
#define LSTR 40   // padded LDS row stride in halfs (80 B = 64 B data + 16 B pad)

__global__ __launch_bounds__(256) void qkv_gemm_wmma(
    const unsigned short* __restrict__ Xb, const unsigned short* __restrict__ Wb,
    float* __restrict__ QKV)
{
  __shared__ unsigned short lA[2][TM * LSTR];
  __shared__ unsigned short lB[2][TN * LSTR];

  const int t      = threadIdx.x;
  const int lane   = t & 31;
  const int wid    = t >> 5;
  const int wave_m = wid & 1;        // 0..1  (64-row strip)
  const int wave_n = wid >> 1;       // 0..3  (32-col strip)
  const int m0     = blockIdx.x * TM;
  const int n0     = blockIdx.y * TN;

  const unsigned ldsA0 = (unsigned)(size_t)&lA[0][0];
  const unsigned ldsA1 = (unsigned)(size_t)&lA[1][0];
  const unsigned ldsB0 = (unsigned)(size_t)&lB[0][0];
  const unsigned ldsB1 = (unsigned)(size_t)&lB[1][0];

  const i32x8 g1A = tdm_group1(B_ * L_);   // 32768 rows
  const i32x8 g1B = tdm_group1(C3_);       // 3072 rows

  const unsigned short* Abase = Xb + (size_t)m0 * DIN_;
  const unsigned short* Bbase = Wb + (size_t)n0 * DIN_;

  v8f acc[4][2] = {};
  const int khalf = lane >> 4;       // 0..1
  const int msub  = lane & 15;

  if (wid == 0) {                    // prologue: DMA panel 0 into buffer 0
    tdm_load_2d(ldsA0, Abase, g1A);
    tdm_load_2d(ldsB0, Bbase, g1B);
  }

  for (int k0 = 0; k0 < DIN_; k0 += TK) {
    const int buf = (k0 >> 5) & 1;
    if (wid == 0) {
      if (k0 + TK < DIN_) {          // DMA next panel into other buffer
        tdm_load_2d(buf ? ldsA0 : ldsA1, Abase + k0 + TK, g1A);
        tdm_load_2d(buf ? ldsB0 : ldsB1, Bbase + k0 + TK, g1B);
        __builtin_amdgcn_s_wait_tensorcnt(2);  // in-order: panel k landed
      } else {
        __builtin_amdgcn_s_wait_tensorcnt(0);
      }
    }
    __syncthreads();                 // panel k visible to all waves

    const unsigned short* La = lA[buf];
    const unsigned short* Lb = lB[buf];

    // ---- A fragments: 16x32 bf16, documented wave32 layout ----
    v16bf afr[4];
#pragma unroll
    for (int tm = 0; tm < 4; ++tm) {
      const int row = wave_m * 64 + tm * 16 + msub;
      const unsigned short* p = La + row * LSTR + khalf * 8;
      v8bf lo = *(const v8bf*)p;            // K = khalf*8 + 0..7
      v8bf hi = *(const v8bf*)(p + 16);     // K = 16 + khalf*8 + 0..7
#pragma unroll
      for (int i = 0; i < 8; ++i) { afr[tm][i] = lo[i]; afr[tm][8 + i] = hi[i]; }
    }
    // ---- B fragments: 32x16 bf16 (lane = N, K split by lane-half) ----
    v16bf bfr[2];
#pragma unroll
    for (int tn = 0; tn < 2; ++tn) {
      const int rowN = wave_n * 32 + tn * 16 + msub;
      const unsigned short* p = Lb + rowN * LSTR + khalf * 8;
      v8bf lo = *(const v8bf*)p;            // K = khalf*8 + 0..7
      v8bf hi = *(const v8bf*)(p + 16);     // K = 16 + khalf*8 + 0..7
#pragma unroll
      for (int i = 0; i < 8; ++i) { bfr[tn][i] = lo[i]; bfr[tn][8 + i] = hi[i]; }
    }

#pragma unroll
    for (int tm = 0; tm < 4; ++tm)
#pragma unroll
      for (int tn = 0; tn < 2; ++tn)
        acc[tm][tn] = __builtin_amdgcn_wmma_f32_16x16x32_bf16(
            false, afr[tm], false, bfr[tn], (short)0, acc[tm][tn], false, false);

    __syncthreads();                 // all waves done reading buf
  }

  // ---- writeback (C/D layout: VGPR r -> M = (lane>>4)*8 + r, N = lane&15) --
  const int mlane = (lane >> 4) * 8;
  const int nlane = lane & 15;
#pragma unroll
  for (int tm = 0; tm < 4; ++tm)
#pragma unroll
    for (int tn = 0; tn < 2; ++tn) {
      const int colg = n0 + wave_n * 32 + tn * 16 + nlane;
      const int rowb = m0 + wave_m * 64 + tm * 16 + mlane;
#pragma unroll
      for (int r = 0; r < 8; ++r)
        QKV[(size_t)(rowb + r) * C3_ + colg] = acc[tm][tn][r];
    }
}

// ===========================================================================
// K2: RoPE(q) + scale + softmax over feature dim (64), in-place in QKV.
//     One wave per (b,l,h) row; lane j owns the rotation pair (j, j+32).
// ===========================================================================
__global__ __launch_bounds__(256) void q_rope_softmax(float* __restrict__ QKV)
{
  const int gid  = blockIdx.x * 8 + (threadIdx.x >> 5);  // (b*L+l)*16 + h
  const int lane = threadIdx.x & 31;
  const int h    = gid & 15;
  float* base = QKV + (size_t)(gid >> 4) * C3_ + h * 192;

  float qlo = base[lane * 3];
  float qhi = base[(lane + 32) * 3];

  const float inv = __expf(-(float)lane * ROPE_LN_SCALE);
  float s, c;
  __sincosf((float)h * inv, &s, &c);
  float rlo = (qlo * c - qhi * s) * 0.125f;   // dh^-0.5 = 1/8
  float rhi = (qhi * c + qlo * s) * 0.125f;

  float mx = fmaxf(rlo, rhi);
#pragma unroll
  for (int o = 16; o > 0; o >>= 1) mx = fmaxf(mx, __shfl_xor(mx, o, 32));
  float elo = __expf(rlo - mx), ehi = __expf(rhi - mx);
  float sm = elo + ehi;
#pragma unroll
  for (int o = 16; o > 0; o >>= 1) sm += __shfl_xor(sm, o, 32);
  const float rinv = 1.0f / sm;

  base[lane * 3]        = elo * rinv;
  base[(lane + 32) * 3] = ehi * rinv;
}

// ===========================================================================
// K3: k sequence-softmax stats: per (b,h,d) max and sum(exp) over the
//     8196 rows (4 un-roped mem rows + 8192 roped/masked rows).
// ===========================================================================
__global__ __launch_bounds__(256) void k_stats(
    const float* __restrict__ QKV, const int* __restrict__ mask,
    const float* __restrict__ memkv,
    float* __restrict__ kmax, float* __restrict__ ksum)
{
  __shared__ float red[256];
  __shared__ float kmx[64];
  const int bh = blockIdx.x, b = bh >> 4, h = bh & 15;
  const int t = threadIdx.x, d = t & 63, rg = t >> 6;

  const int   j   = d & 31;
  const float inv = __expf(-(float)j * ROPE_LN_SCALE);
  float s, c;
  __sincosf((float)h * inv, &s, &c);
  const int   pair = (d < 32) ? d + 32 : d - 32;
  const float sgn  = (d < 32) ? -1.0f : 1.0f;   // k' = k[d]*c + sgn*k[pair]*s

  const float* qbase = QKV + (size_t)b * L_ * C3_ + h * 192;
  const int*   mrow  = mask + b * L_;
  const float  memv  = memkv[(h * NMEM_ + rg) * DH_ + d];

  float lmax = memv;
  for (int l = rg; l < L_; l += 4) {
    if (mrow[l]) {
      const float* p = qbase + (size_t)l * C3_;
      float kp = p[d * 3 + 1] * c + sgn * p[pair * 3 + 1] * s;
      lmax = fmaxf(lmax, kp);
    }
  }
  red[t] = lmax;
  __syncthreads();
  if (t < 64) {
    float m = fmaxf(fmaxf(red[d], red[64 + d]), fmaxf(red[128 + d], red[192 + d]));
    kmx[d] = m;
    kmax[bh * 64 + d] = m;
  }
  __syncthreads();

  const float M = kmx[d];
  float lsum = __expf(memv - M);
  for (int l = rg; l < L_; l += 4) {
    if (mrow[l]) {
      const float* p = qbase + (size_t)l * C3_;
      float kp = p[d * 3 + 1] * c + sgn * p[pair * 3 + 1] * s;
      lsum += __expf(kp - M);
    }
  }
  __syncthreads();
  red[t] = lsum;
  __syncthreads();
  if (t < 64)
    ksum[bh * 64 + d] = red[d] + red[64 + d] + red[128 + d] + red[192 + d];
}

// ===========================================================================
// K4: ctx[b,h,d,e] = sum_n softmax_k[n,d] * v[n,e]   (64x64 per (b,h))
//     8-row LDS panels; thread owns (d, 16 e's).
// ===========================================================================
__global__ __launch_bounds__(256) void ctx_kernel(
    const float* __restrict__ QKV, const int* __restrict__ mask,
    const float* __restrict__ memkv,
    const float* __restrict__ kmax, const float* __restrict__ ksum,
    float* __restrict__ CTX)
{
  __shared__ float kex[8][64];
  __shared__ float vrw[8][64];
  __shared__ float kmx[64], rks[64];

  const int bh = blockIdx.x, b = bh >> 4, h = bh & 15;
  const int t = threadIdx.x;
  if (t < 64) {
    kmx[t] = kmax[bh * 64 + t];
    rks[t] = 1.0f / ksum[bh * 64 + t];
  }
  __syncthreads();

  // loader identity (dd is the same for both half-loads since 256 % 64 == 0)
  const int dd  = t & 63;
  const int rr0 = t >> 6;
  const int   jj   = dd & 31;
  const float invf = __expf(-(float)jj * ROPE_LN_SCALE);
  float ss, cc;
  __sincosf((float)h * invf, &ss, &cc);
  const int   pair = (dd < 32) ? dd + 32 : dd - 32;
  const float sg   = (dd < 32) ? -1.0f : 1.0f;
  const float km   = kmx[dd];
  const float rk   = rks[dd];

  const float* qbase = QKV + (size_t)b * L_ * C3_ + h * 192;
  const int*   mrow  = mask + b * L_;

  const int dwn = t >> 2;            // accumulator d
  const int e0  = (t & 3) << 4;      // accumulator e base (16 e's)
  float acc[16] = {};

  for (int n0 = 0; n0 < NSEQ_; n0 += 8) {
#pragma unroll
    for (int ii = 0; ii < 2; ++ii) {
      const int rr = rr0 + ii * 4;
      const int n  = n0 + rr;
      float kv = 0.0f, vv = 0.0f;
      if (n < NSEQ_) {
        if (n < NMEM_) {
          kv = __expf(memkv[(h * NMEM_ + n) * DH_ + dd] - km) * rk;
          vv = memkv[H_ * NMEM_ * DH_ + (h * NMEM_ + n) * DH_ + dd];
        } else {
          const int l = n - NMEM_;
          const float* p = qbase + (size_t)l * C3_;
          vv = p[dd * 3 + 2];
          if (mrow[l]) {
            float kp = p[dd * 3 + 1] * cc + sg * p[pair * 3 + 1] * ss;
            kv = __expf(kp - km) * rk;
          }
        }
      }
      kex[rr][dd] = kv;
      vrw[rr][dd] = vv;
    }
    __syncthreads();
#pragma unroll
    for (int r = 0; r < 8; ++r) {
      const float sk = kex[r][dwn];
#pragma unroll
      for (int q = 0; q < 16; ++q) acc[q] += sk * vrw[r][e0 + q];
    }
    __syncthreads();
  }

  float* cb = CTX + ((size_t)bh * 64 + dwn) * 64 + e0;
#pragma unroll
  for (int q = 0; q < 16; ++q) cb[q] = acc[q];
}

// ===========================================================================
// K5: out[b,l,h*64+e] = mask * sum_d q_sm[b,l,h,d] * ctx[b,h,d,e]
//     Block = (b, h, 4 consecutive l rows); ctx tile in LDS.
// ===========================================================================
__global__ __launch_bounds__(256) void out_kernel(
    const float* __restrict__ QKV, const int* __restrict__ mask,
    const float* __restrict__ CTX, float* __restrict__ out)
{
  __shared__ float cs[64 * 64];
  __shared__ float qs[4 * 64];

  const int b = blockIdx.z, h = blockIdx.y;
  const int l0 = blockIdx.x * 4;
  const int t = threadIdx.x;

  const float* cbase = CTX + (size_t)(b * H_ + h) * (DH_ * DH_);
#pragma unroll
  for (int i = 0; i < 16; ++i) cs[t + i * 256] = cbase[t + i * 256];

  const int lr = t >> 6, e = t & 63;
  qs[t] = QKV[(size_t)(b * L_ + l0 + lr) * C3_ + h * 192 + (t & 63) * 3];
  __syncthreads();

  float acc = 0.0f;
#pragma unroll 8
  for (int d = 0; d < 64; ++d) acc += qs[lr * 64 + d] * cs[d * 64 + e];

  const int m = mask[b * L_ + l0 + lr];
  out[(size_t)(b * L_ + l0 + lr) * 1024 + h * DH_ + e] = m ? acc : 0.0f;
}

// ===========================================================================
extern "C" void kernel_launch(void* const* d_in, const int* in_sizes, int n_in,
                              void* d_out, int out_size, void* d_ws, size_t ws_size,
                              hipStream_t stream) {
  (void)in_sizes; (void)n_in; (void)out_size; (void)ws_size;

  const float* x     = (const float*)d_in[0];
  const int*   mask  = (const int*)d_in[1];
  const float* W     = (const float*)d_in[2];
  const float* memkv = (const float*)d_in[3];
  float*       out   = (float*)d_out;

  char* ws = (char*)d_ws;
  const size_t XB_BYTES  = (size_t)B_ * L_ * DIN_ * 2;   //  67,108,864
  const size_t WB_BYTES  = (size_t)C3_ * DIN_ * 2;       //   6,291,456
  const size_t QKV_BYTES = (size_t)B_ * L_ * C3_ * 4;    // 402,653,184

  unsigned short* Xb  = (unsigned short*)ws;
  unsigned short* Wb  = (unsigned short*)(ws + XB_BYTES);
  float* QKV   = (float*)(ws + XB_BYTES + WB_BYTES);
  float* kmaxp = (float*)(ws + XB_BYTES + WB_BYTES + QKV_BYTES);
  float* ksump = (float*)(ws + XB_BYTES + WB_BYTES + QKV_BYTES + 16384);
  float* CTX   = (float*)(ws + XB_BYTES + WB_BYTES + QKV_BYTES + 32768);

  cvt_bf16<<<(B_ * L_ * DIN_) / 2048, 256, 0, stream>>>(x, Xb);
  cvt_bf16<<<(C3_ * DIN_) / 2048, 256, 0, stream>>>(W, Wb);
  qkv_gemm_wmma<<<dim3((B_ * L_) / TM, C3_ / TN), 256, 0, stream>>>(Xb, Wb, QKV);
  q_rope_softmax<<<(B_ * L_ * H_) / 8, 256, 0, stream>>>(QKV);
  k_stats<<<B_ * H_, 256, 0, stream>>>(QKV, mask, memkv, kmaxp, ksump);
  ctx_kernel<<<B_ * H_, 256, 0, stream>>>(QKV, mask, memkv, kmaxp, ksump, CTX);
  out_kernel<<<dim3(L_ / 4, H_, B_), 256, 0, stream>>>(QKV, mask, CTX, out);
}
